// SelfAttention_13357348290569
// MI455X (gfx1250) — matrix operand order repaired and checked by reading, not compiled, not run on hardware
//
#include <hip/hip_runtime.h>
#include <hip/hip_bf16.h>
#include <math.h>

// ---------------------------------------------------------------------------
// Problem constants (from reference)
// ---------------------------------------------------------------------------
#define BATCH 4
#define SEQ   2048
#define DMODEL 1024          // EMBED_DIM == KEY_DIM
#define E3    (3 * DMODEL)   // 3072
#define MROWS (BATCH * SEQ)  // 8192

typedef __bf16 v16bf __attribute__((ext_vector_type(16)));
typedef __bf16 v8bf  __attribute__((ext_vector_type(8)));
typedef float  v8f   __attribute__((ext_vector_type(8)));

// Pointer types for the async global->LDS builtin (prototype revealed by the
// round-2 diagnostic: v4i addrspace(1)* / v4i addrspace(3)*).
typedef int v4i_vs __attribute__((vector_size(16)));
typedef __attribute__((address_space(1))) v4i_vs* gv4i_p;
typedef __attribute__((address_space(3))) v4i_vs* lv4i_p;

// ---------------------------------------------------------------------------
// CDNA5 async global->LDS path (probe-guarded; sync fallback otherwise)
// ---------------------------------------------------------------------------
#if defined(__has_builtin)
#  if __has_builtin(__builtin_amdgcn_global_load_async_to_lds_b128)
#    define USE_ASYNC_LDS 1
#  else
#    define USE_ASYNC_LDS 0
#  endif
#else
#  define USE_ASYNC_LDS 0
#endif

#if USE_ASYNC_LDS
#  if __has_builtin(__builtin_amdgcn_s_wait_asynccnt)
#    define WAIT_ASYNC0() __builtin_amdgcn_s_wait_asynccnt(0)
#  else
#    define WAIT_ASYNC0() asm volatile("s_wait_asynccnt 0" ::: "memory")
#  endif
#else
#  define WAIT_ASYNC0()
#endif

// ---------------------------------------------------------------------------
// fp32 -> bf16 elementwise convert (x)
// ---------------------------------------------------------------------------
__global__ __launch_bounds__(256)
void convert_f32_bf16(const float* __restrict__ src, __bf16* __restrict__ dst, int n) {
    int i = (blockIdx.x * 256 + threadIdx.x) * 4;
    if (i + 3 < n) {
        float4 v = *(const float4*)(src + i);
        dst[i + 0] = (__bf16)v.x;
        dst[i + 1] = (__bf16)v.y;
        dst[i + 2] = (__bf16)v.z;
        dst[i + 3] = (__bf16)v.w;
    }
}

// ---------------------------------------------------------------------------
// W (1024 x 3072) fp32  ->  Wt (3072 x 1024) bf16, LDS-tiled transpose
// ---------------------------------------------------------------------------
__global__ __launch_bounds__(256)
void transpose_W_bf16(const float* __restrict__ W, __bf16* __restrict__ Wt) {
    __shared__ float tile[32][33];
    const int e0 = blockIdx.x * 32;  // along 3072
    const int d0 = blockIdx.y * 32;  // along 1024
    const int tx = threadIdx.x;      // 0..31
    const int ty = threadIdx.y;      // 0..7
#pragma unroll
    for (int j = 0; j < 4; ++j)
        tile[ty + j * 8][tx] = W[(size_t)(d0 + ty + j * 8) * E3 + (e0 + tx)];
    __syncthreads();
#pragma unroll
    for (int j = 0; j < 4; ++j)
        Wt[(size_t)(e0 + ty + j * 8) * DMODEL + (d0 + tx)] =
            (__bf16)tile[tx][ty + j * 8];
}

// ---------------------------------------------------------------------------
// Column-wise online softmax over q (axis=1 of (b,q,k)); bf16 output
// ---------------------------------------------------------------------------
__global__ __launch_bounds__(256)
void softmax_cols(const float* __restrict__ scores, __bf16* __restrict__ attn) {
    const int k = blockIdx.x * 256 + threadIdx.x;   // column
    const int b = blockIdx.y;
    const float* sc = scores + (size_t)b * SEQ * SEQ;
    __bf16* at = attn + (size_t)b * SEQ * SEQ;

    float m = -INFINITY, l = 0.0f;
    for (int q = 0; q < SEQ; ++q) {
        float v = sc[(size_t)q * SEQ + k];
        float nm = fmaxf(m, v);
        l = l * __expf(m - nm) + __expf(v - nm);
        m = nm;
    }
    float inv = 1.0f / l;
    for (int q = 0; q < SEQ; ++q) {
        float v = sc[(size_t)q * SEQ + k];
        at[(size_t)q * SEQ + k] = (__bf16)(__expf(v - m) * inv);
    }
}

// ---------------------------------------------------------------------------
// Generic bf16 WMMA GEMM.  C(M,N) = A(M,K) * B(N,K)^T   (B stored N-major,
// contiguous over K == exactly what the v_wmma B-fragment layout wants).
//
// Block = 128 threads (4 waves), wave grid 2x2, each wave 64x64 output via
// 4x4 v_wmma_f32_16x16x32_bf16 accumulators.  Double-buffered LDS staging:
// async global->LDS loads (ASYNCcnt) of tile k+1 overlap WMMA on tile k,
// one barrier per K-step.
// ---------------------------------------------------------------------------
#define LDS_STRIDE 40   // 32 elems + 8 pad (bank-conflict spread)

static __device__ __forceinline__ v16bf ld_frag(const __bf16* p0, const __bf16* p1) {
    v8bf lo = *(const v8bf*)p0;
    v8bf hi = *(const v8bf*)p1;
    v16bf r;
#pragma unroll
    for (int i = 0; i < 8; ++i) { r[i] = lo[i]; r[i + 8] = hi[i]; }
    return r;
}

static __device__ __forceinline__ void stage_tiles(
    const __bf16* __restrict__ A, const __bf16* __restrict__ B,
    __bf16* lA, __bf16* lB, int tid, int lda, int ldb, int bM, int bN, int k0) {
#pragma unroll
    for (int c = 0; c < 4; ++c) {
        int idx = tid + c * 128;
        int r   = idx >> 2;
        int col = (idx & 3) * 8;
        const __bf16* ga = A + (size_t)(bM + r) * lda + k0 + col;
        const __bf16* gb = B + (size_t)(bN + r) * ldb + k0 + col;
        __bf16* la = lA + r * LDS_STRIDE + col;
        __bf16* lb = lB + r * LDS_STRIDE + col;
#if USE_ASYNC_LDS
        __builtin_amdgcn_global_load_async_to_lds_b128((gv4i_p)ga, (lv4i_p)la, 0, 0);
        __builtin_amdgcn_global_load_async_to_lds_b128((gv4i_p)gb, (lv4i_p)lb, 0, 0);
#else
        *(v8bf*)la = *(const v8bf*)ga;
        *(v8bf*)lb = *(const v8bf*)gb;
#endif
    }
}

template <int MODE>
__global__ __launch_bounds__(128)
void gemm_bf16_wmma(const __bf16* __restrict__ Abase,
                    const __bf16* __restrict__ Bbase,
                    __bf16* __restrict__ q_out,
                    __bf16* __restrict__ k_out,
                    __bf16* __restrict__ vt_out,
                    float*  __restrict__ f_out) {
    constexpr int Kdim = (MODE == 2) ? 2048 : 1024;
    constexpr int lda  = (MODE == 2) ? 2048 : 1024;
    constexpr int ldb  = (MODE == 2) ? 2048 : 1024;

    const int z = blockIdx.z;
    const __bf16* A = Abase;
    const __bf16* B = Bbase;
    if (MODE == 1) {
        A += (size_t)z * SEQ * DMODEL;
        B += (size_t)z * SEQ * DMODEL;
    } else if (MODE == 2) {
        A += (size_t)z * SEQ * SEQ;
        B += (size_t)z * DMODEL * SEQ;
    }

    __shared__ __bf16 Alds[2][128 * LDS_STRIDE];
    __shared__ __bf16 Blds[2][128 * LDS_STRIDE];

    const int tid  = threadIdx.x;
    const int lane = tid & 31;
    const int wid  = tid >> 5;
    const int wm   = wid >> 1;          // wave row  (0..1)
    const int wn   = wid & 1;           // wave col  (0..1)
    const int hl   = lane >> 4;         // lane half (0..1)
    const int ln   = lane & 15;

    const int bM = blockIdx.y * 128;
    const int bN = blockIdx.x * 128;

    v8f acc[4][4] = {};

    // Prologue: stage first K-tile into buffer 0.
    stage_tiles(A, B, Alds[0], Blds[0], tid, lda, ldb, bM, bN, 0);

    int cur = 0;
    for (int k0 = 0; k0 < Kdim; k0 += 32) {
        WAIT_ASYNC0();        // my async writes into buf[cur] have landed
        __syncthreads();      // everyone's writes landed; prior reads of buf[cur^1] done

        if (k0 + 32 < Kdim)   // prefetch next K-tile into the other buffer
            stage_tiles(A, B, Alds[cur ^ 1], Blds[cur ^ 1], tid, lda, ldb, bM, bN, k0 + 32);

        // Fragment loads per documented VGPR layouts.
        const __bf16* Acur = Alds[cur];
        const __bf16* Bcur = Blds[cur];
        v16bf afr[4], bfr[4];
#pragma unroll
        for (int i = 0; i < 4; ++i) {
            // A 16x32: lane row = ln; K chunks at hl*8 and hl*8+16
            const __bf16* ap = Acur + (wm * 64 + i * 16 + ln) * LDS_STRIDE + hl * 8;
            afr[i] = ld_frag(ap, ap + 16);
            // B 32x16: lane col = ln; K chunks at hl*8 and hl*8+16
            const __bf16* bp = Bcur + (wn * 64 + i * 16 + ln) * LDS_STRIDE + hl * 8;
            bfr[i] = ld_frag(bp, bp + 16);
        }
#pragma unroll
        for (int i = 0; i < 4; ++i)
#pragma unroll
            for (int j = 0; j < 4; ++j)
                acc[i][j] = __builtin_amdgcn_wmma_f32_16x16x32_bf16(
                    false, afr[i], false, bfr[j], (short)0, acc[i][j], false, false);
        cur ^= 1;
    }

    // ------------------------------------------------------------------
    // Epilogue. C/D layout: VGPR r -> M = r + hl*8, N = ln.
    // ------------------------------------------------------------------
    if (MODE == 0) {
        // Region is uniform per block: N-tiles are 128-wide, boundaries at 1024/2048.
        const int b = bM >> 11;                 // tile never crosses a batch (2048 | 128)
        if (bN < 2 * DMODEL) {
            __bf16* dst = (bN < DMODEL) ? q_out : k_out;
            const int nOff = (bN < DMODEL) ? 0 : DMODEL;
#pragma unroll
            for (int i = 0; i < 4; ++i)
#pragma unroll
                for (int j = 0; j < 4; ++j)
#pragma unroll
                    for (int r = 0; r < 8; ++r) {
                        const int s  = (bM + wm * 64 + i * 16 + hl * 8 + r) & (SEQ - 1);
                        const int gn = bN + wn * 64 + j * 16 + ln - nOff;
                        dst[((size_t)b * SEQ + s) * DMODEL + gn] = (__bf16)acc[i][j][r];
                    }
        } else {
            // V region: 8 consecutive M-values per lane are 8 consecutive s ->
            // pack into one b128 store per accumulator tile.
#pragma unroll
            for (int i = 0; i < 4; ++i)
#pragma unroll
                for (int j = 0; j < 4; ++j) {
                    const int s0 = (bM + wm * 64 + i * 16 + hl * 8) & (SEQ - 1);
                    const int d  = bN + wn * 64 + j * 16 + ln - 2 * DMODEL;
                    v8bf pk;
#pragma unroll
                    for (int r = 0; r < 8; ++r) pk[r] = (__bf16)acc[i][j][r];
                    *(v8bf*)(vt_out + ((size_t)b * DMODEL + d) * SEQ + s0) = pk;
                }
        }
    } else {
#pragma unroll
        for (int i = 0; i < 4; ++i)
#pragma unroll
            for (int j = 0; j < 4; ++j)
#pragma unroll
                for (int r = 0; r < 8; ++r) {
                    const int gm = bM + wm * 64 + i * 16 + hl * 8 + r;
                    const int gn = bN + wn * 64 + j * 16 + ln;
                    if (MODE == 1) {
                        float* dst = f_out + (size_t)z * SEQ * SEQ;
                        dst[(size_t)gm * SEQ + gn] = acc[i][j][r] * 0.03125f; // 1/sqrt(1024)
                    } else {
                        float* dst = f_out + (size_t)z * SEQ * DMODEL;
                        dst[(size_t)gm * DMODEL + gn] = acc[i][j][r];
                    }
                }
    }
}

// ---------------------------------------------------------------------------
// Host-side launch
// ---------------------------------------------------------------------------
extern "C" void kernel_launch(void* const* d_in, const int* in_sizes, int n_in,
                              void* d_out, int out_size, void* d_ws, size_t ws_size,
                              hipStream_t stream) {
    const float* x = (const float*)d_in[0];   // (4, 2048, 1024)
    const float* W = (const float*)d_in[1];   // (1024, 3072)
    float* out = (float*)d_out;               // (4, 2048, 1024)

    char* ws = (char*)d_ws;
    const size_t MB = 1024ull * 1024ull;
    __bf16* xbf    = (__bf16*)(ws + 0);          // 16 MB : (8192, 1024)
    __bf16* Wt     = (__bf16*)(ws + 16 * MB);    //  6 MB : (3072, 1024)
    __bf16* Qb     = (__bf16*)(ws + 22 * MB);    // 16 MB : (4, 2048, 1024)
    __bf16* Kb     = (__bf16*)(ws + 38 * MB);    // 16 MB : (4, 2048, 1024)
    __bf16* Vt     = (__bf16*)(ws + 54 * MB);    // 16 MB : (4, 1024, 2048)
    float*  scores = (float*) (ws + 70 * MB);    // 64 MB : (4, 2048, 2048)
    __bf16* attn   = (__bf16*)(ws + 134 * MB);   // 32 MB : (4, 2048, 2048)

    // 1) downconvert inputs to bf16 (W transposed so GEMM B-columns are contiguous)
    convert_f32_bf16<<<dim3(MROWS * DMODEL / (256 * 4)), dim3(256), 0, stream>>>(
        x, xbf, MROWS * DMODEL);
    transpose_W_bf16<<<dim3(E3 / 32, DMODEL / 32), dim3(32, 8), 0, stream>>>(W, Wt);

    // 2) QKV projection: (8192x1024) @ (1024x3072)
    gemm_bf16_wmma<0><<<dim3(E3 / 128, MROWS / 128, 1), dim3(128), 0, stream>>>(
        xbf, Wt, Qb, Kb, Vt, nullptr);

    // 3) scores = Q K^T / sqrt(d), per batch
    gemm_bf16_wmma<1><<<dim3(SEQ / 128, SEQ / 128, BATCH), dim3(128), 0, stream>>>(
        Qb, Kb, nullptr, nullptr, nullptr, scores);

    // 4) softmax over the query axis (axis=1 of (b,q,k))
    softmax_cols<<<dim3(SEQ / 256, BATCH), dim3(256), 0, stream>>>(scores, attn);

    // 5) out = attn @ V  (V stored transposed: B N-major over d, contiguous over s)
    gemm_bf16_wmma<2><<<dim3(DMODEL / 128, SEQ / 128, BATCH), dim3(128), 0, stream>>>(
        attn, Vt, nullptr, nullptr, nullptr, out);
}
